// KernalAnsatz_65481071409897
// MI455X (gfx1250) — matrix operand order repaired and checked by reading, not compiled, not run on hardware
//
#include <hip/hip_runtime.h>

typedef __attribute__((ext_vector_type(16))) __bf16 v16bf;
typedef __attribute__((ext_vector_type(8)))  float  v8f;

#define NROWS 8192
#define MROWS 8192
#define DDIM  128
#define RSTRIDE 256   // packed row stride: 128 hi + 128 lo bf16

// ---------------------------------------------------------------------------
// Kernel 1: g = softplus(gamma), 128 elements
// ---------------------------------------------------------------------------
__global__ void softplus_kernel(const float* __restrict__ gamma, float* __restrict__ g) {
    int d = threadIdx.x;
    float v = gamma[d];
    g[d] = fmaxf(v, 0.0f) + log1pf(__expf(-fabsf(v)));
}

// ---------------------------------------------------------------------------
// Kernel 2: build packed bf16 hi/lo rows and weighted row norms.
//   Apk row n: [ bf16(x*g) x128 | bf16 residual x128 ],  x2[n] = sum g x^2
//   Bpk row m: [ bf16(y)   x128 | bf16 residual x128 ],  y2[m] = sum g y^2
// ---------------------------------------------------------------------------
__global__ void prep_rows(const float* __restrict__ x, const float* __restrict__ y,
                          const float* __restrict__ g,
                          __bf16* __restrict__ Apk, __bf16* __restrict__ Bpk,
                          float* __restrict__ x2, float* __restrict__ y2) {
    __shared__ float red[128];
    const int d   = threadIdx.x;
    const int row = blockIdx.x;
    const float gd = g[d];
    float partial;
    if (row < NROWS) {
        float xv = x[(size_t)row * DDIM + d];
        float a  = xv * gd;
        __bf16 hi = (__bf16)a;
        __bf16 lo = (__bf16)(a - (float)hi);
        Apk[(size_t)row * RSTRIDE + d]        = hi;
        Apk[(size_t)row * RSTRIDE + 128 + d]  = lo;
        partial = a * xv;                      // g * x^2 in exact f32
    } else {
        int m = row - NROWS;
        float yv = y[(size_t)m * DDIM + d];
        __bf16 hi = (__bf16)yv;
        __bf16 lo = (__bf16)(yv - (float)hi);
        Bpk[(size_t)m * RSTRIDE + d]          = hi;
        Bpk[(size_t)m * RSTRIDE + 128 + d]    = lo;
        partial = gd * yv * yv;
    }
    red[d] = partial;
    __syncthreads();
    for (int s = 64; s > 0; s >>= 1) {
        if (d < s) red[d] += red[d + s];
        __syncthreads();
    }
    if (d == 0) {
        if (row < NROWS) x2[row] = red[0];
        else             y2[row - NROWS] = red[0];
    }
}

// ---------------------------------------------------------------------------
// Kernel 3: tiled split-bf16 WMMA GEMM + fused exp epilogue.
// Workgroup (8 waves) -> 128x128 output tile; wave -> 16 rows x 128 cols.
// k-loop kept rolled to bound VGPR pressure (<256) and keep occupancy.
// ---------------------------------------------------------------------------
__global__ __launch_bounds__(256, 2) void rbf_wmma_kernel(
    const __bf16* __restrict__ Apk, const __bf16* __restrict__ Bpk,
    const float*  __restrict__ x2,  const float*  __restrict__ y2,
    float* __restrict__ out)
{
    const int lane  = threadIdx.x & 31;
    const int wave  = threadIdx.x >> 5;
    const int n0    = blockIdx.y * 128 + wave * 16;  // 16 rows for this wave
    const int m0    = blockIdx.x * 128;              // 128 cols for workgroup
    const int lhalf = lane >> 4;                     // 0 or 1
    const int lmod  = lane & 15;

    union F { v16bf v; uint4 q[2]; };

    v8f zero = {};
    v8f acc[8];
#pragma unroll
    for (int t = 0; t < 8; ++t) acc[t] = zero;

    // 32-bit element offsets (GVS addressing: SGPR base + VGPR offset)
    const int a_off = (n0 + lmod) * RSTRIDE + lhalf * 8;   // A: row M=lane&15
    const int b_off = (m0 + lmod) * RSTRIDE + lhalf * 8;   // B: col N=lane&15

#pragma unroll 1
    for (int kk = 0; kk < DDIM; kk += 32) {
        // A fragment: K = 8*lhalf+{0..7}, 16+8*lhalf+{0..7}; lo copy at +128 elems
        F fah, fal;
        const int ao = a_off + kk;
        fah.q[0] = *(const uint4*)(Apk + ao);
        fah.q[1] = *(const uint4*)(Apk + ao + 16);
        fal.q[0] = *(const uint4*)(Apk + ao + 128);
        fal.q[1] = *(const uint4*)(Apk + ao + 144);

#pragma unroll
        for (int t = 0; t < 8; ++t) {
            // B fragment: K = 8*lhalf+{0..7}, 16+8*lhalf+{0..7}; lo copy at +128 elems
            const int bo = b_off + t * (16 * RSTRIDE) + kk;
            F fbh, fbl;
            fbh.q[0] = *(const uint4*)(Bpk + bo);
            fbh.q[1] = *(const uint4*)(Bpk + bo + 16);
            fbl.q[0] = *(const uint4*)(Bpk + bo + 128);
            fbl.q[1] = *(const uint4*)(Bpk + bo + 144);

            // split product: A*B ~= Ah*Bh + Ah*Bl + Al*Bh   (f32 accumulate)
            acc[t] = __builtin_amdgcn_wmma_f32_16x16x32_bf16(
                false, fah.v, false, fbh.v, (short)0, acc[t], false, false);
            acc[t] = __builtin_amdgcn_wmma_f32_16x16x32_bf16(
                false, fah.v, false, fbl.v, (short)0, acc[t], false, false);
            acc[t] = __builtin_amdgcn_wmma_f32_16x16x32_bf16(
                false, fal.v, false, fbh.v, (short)0, acc[t], false, false);
        }
    }

    // Epilogue: D layout -> VGPR r holds (M = r + 8*lhalf, N = lane&15)
    const int nb = n0 + lhalf * 8;
    float4 xa = *(const float4*)(x2 + nb);
    float4 xb = *(const float4*)(x2 + nb + 4);
    float xr[8] = {xa.x, xa.y, xa.z, xa.w, xb.x, xb.y, xb.z, xb.w};

#pragma unroll
    for (int t = 0; t < 8; ++t) {
        const int m  = m0 + t * 16 + lmod;
        const float y2v = y2[m];
        float* orow = out + (size_t)nb * MROWS + m;
#pragma unroll
        for (int r = 0; r < 8; ++r) {
            float sq = xr[r] + y2v - 2.0f * acc[t][r];
            orow[(size_t)r * MROWS] = __expf(-sq);
        }
    }
}

// ---------------------------------------------------------------------------
// Launch
// ---------------------------------------------------------------------------
extern "C" void kernel_launch(void* const* d_in, const int* in_sizes, int n_in,
                              void* d_out, int out_size, void* d_ws, size_t ws_size,
                              hipStream_t stream) {
    const float* x     = (const float*)d_in[0];
    const float* y     = (const float*)d_in[1];
    const float* gamma = (const float*)d_in[2];
    float* out = (float*)d_out;

    // Workspace layout (~8.1 MB total)
    float* g  = (float*)d_ws;                 // 128
    float* x2 = g + 128;                      // 8192
    float* y2 = x2 + NROWS;                   // 8192
    __bf16* Apk = (__bf16*)(y2 + MROWS);      // 8192*256 bf16 (hi|lo packed)
    __bf16* Bpk = Apk + (size_t)NROWS * RSTRIDE;

    softplus_kernel<<<1, 128, 0, stream>>>(gamma, g);
    prep_rows<<<NROWS + MROWS, 128, 0, stream>>>(x, y, g, Apk, Bpk, x2, y2);
    rbf_wmma_kernel<<<dim3(MROWS / 128, NROWS / 128), 256, 0, stream>>>(
        Apk, Bpk, x2, y2, out);
}
